// SelfAttention_489626272173
// MI455X (gfx1250) — matrix operand order repaired and checked by reading, not compiled, not run on hardware
//
#include <hip/hip_runtime.h>

// ---------------------------------------------------------------------------
// Self-attention (B=8, S=2048, D=1024) for gfx1250 / MI455X.
// Compute-bound (~240 GFLOP vs ~140MB traffic) -> all GEMMs via
// v_wmma_f32_16x16x32_bf16 (fp32 accumulate); operand staging via the
// Tensor Data Mover (tensor_load_to_lds, TENSORcnt) with LDS double
// buffering; softmax in fp32. 1/sqrt(D) folded into Q.
// ---------------------------------------------------------------------------

typedef __bf16 bf16;
typedef __attribute__((ext_vector_type(4)))  __bf16 v4bf;
typedef __attribute__((ext_vector_type(8)))  __bf16 v8bf;
typedef __attribute__((ext_vector_type(16))) __bf16 v16bf;
typedef __attribute__((ext_vector_type(8)))  float  v8f;
typedef __attribute__((ext_vector_type(4)))  float  v4f;
typedef unsigned int u32x4 __attribute__((ext_vector_type(4)));
typedef int          i32x4 __attribute__((ext_vector_type(4)));
typedef int          i32x8 __attribute__((ext_vector_type(8)));

#define B_  8
#define S_  2048
#define D_  1024
#define LDK 48  // padded LDS row stride (bf16): 64B data + 32B pad = 96B (16B aligned)

// ---------------- TDM: DMA a 128-row x 32-col bf16 tile into LDS -----------
// D# per ISA cdna5 §8.3/8.4. data_size=2B; tile_dim0=32 (64B rows = 16 DWORDs),
// pad_interval=3 (16 DWORDs), pad_amount=7 (8 DWORDs = 32B) -> LDS row stride
// 96B == LDK*2. Tensor dims set huge (tiles are always fully in-bounds here).
__device__ __forceinline__ void tdm_load_tile_128x32(unsigned lds_byte_addr,
                                                     const bf16* gptr,
                                                     int stride_elems) {
  unsigned long long ga = (unsigned long long)(const void*)gptr;
  u32x4 g0;
  g0[0] = 1u;                                                  // count=1
  g0[1] = lds_byte_addr;                                       // lds_addr
  g0[2] = (unsigned)ga;                                        // global_addr[31:0]
  g0[3] = (unsigned)((ga >> 32) & 0x01FFFFFFu) | (2u << 30);   // addr[56:32], type=2
  const unsigned td0 = 1u << 20, td1 = 1u << 20;
  i32x8 g1;
  g1[0] = (int)((1u << 16)      // data_size = 2 bytes
              | (1u << 20)      // pad_enable
              | (3u << 22)      // pad_interval = 16 DWORDs
              | (7u << 25));    // pad_amount   = 8 DWORDs
  g1[1] = (int)((td0 & 0xFFFFu) << 16);                        // tensor_dim0[15:0]
  g1[2] = (int)((td0 >> 16) | ((td1 & 0xFFFFu) << 16));        // dim0[31:16], dim1[15:0]
  g1[3] = (int)((td1 >> 16) | (32u << 16));                    // dim1[31:16], tile_dim0=32
  g1[4] = (int)128u;                                           // tile_dim1=128, tile_dim2=0
  g1[5] = (int)(unsigned)stride_elems;                         // tensor_dim0_stride[31:0]
  g1[6] = 0;
  g1[7] = 0;
  const i32x4 gz = {0, 0, 0, 0};
#if defined(__clang_major__) && (__clang_major__ >= 23)
  const i32x8 gz8 = {0, 0, 0, 0, 0, 0, 0, 0};
  __builtin_amdgcn_tensor_load_to_lds(g0, g1, gz, gz, gz8, 0);
#else
  __builtin_amdgcn_tensor_load_to_lds(g0, g1, gz, gz, 0);
#endif
}

// ---------------- fragment assembly (ISA §7.12.2 layouts) ------------------
// A 16x32 bf16: lane half h, row r16; elems 0..7 -> K=8h+0..7, 8..15 -> K=16+8h+0..7
__device__ __forceinline__ v16bf frag_a(const bf16* tile, int row, int half) {
  const bf16* p = tile + row * LDK;
  v8bf lo = *(const v8bf*)(p + half * 8);
  v8bf hi = *(const v8bf*)(p + 16 + half * 8);
  return __builtin_shufflevector(lo, hi, 0, 1, 2, 3, 4, 5, 6, 7, 8, 9, 10, 11, 12, 13, 14, 15);
}
// B 32x16 bf16 (stored as [n][k]): lane = col n; elems 0..7 -> K=8h+0..7, 8..15 -> K=16+8h+0..7
__device__ __forceinline__ v16bf frag_b(const bf16* tile, int nrow, int half) {
  const bf16* p = tile + nrow * LDK;
  v8bf lo = *(const v8bf*)(p + half * 8);
  v8bf hi = *(const v8bf*)(p + 16 + half * 8);
  return __builtin_shufflevector(lo, hi, 0, 1, 2, 3, 4, 5, 6, 7, 8, 9, 10, 11, 12, 13, 14, 15);
}

// Epilogue: D layout — lane n = r16; VGPR r holds row r + 8*half
template <bool OUT_BF16>
__device__ __forceinline__ void epilogue(v8f (&acc)[2][4], void* C, int ldc, float scale,
                                         int bm, int bn, int wm, int wn, int half, int r16) {
#pragma unroll
  for (int i = 0; i < 2; i++)
#pragma unroll
    for (int j = 0; j < 4; j++) {
      const int col = bn + wn + j * 16 + r16;
#pragma unroll
      for (int r = 0; r < 8; r++) {
        const int row = bm + wm + i * 16 + r + 8 * half;
        const float v = acc[i][j][r] * scale;
        if (OUT_BF16) ((bf16*)C)[(size_t)row * ldc + col] = (bf16)v;
        else          ((float*)C)[(size_t)row * ldc + col] = v;
      }
    }
}

// ---------------- GEMM core, TDM double-buffered ---------------------------
// C[128x128 tile] = A[MxK] * B, with Bt = B^T given row-major [N x K].
template <bool OUT_BF16>
__device__ __forceinline__ void gemm_core_tdm(const bf16* __restrict__ A, int lda,
                                              const bf16* __restrict__ Bt, int ldb,
                                              void* __restrict__ C, int ldc, int Kdim,
                                              float scale, int bm, int bn) {
  __shared__ __align__(16) bf16 As[2][128 * LDK];
  __shared__ __align__(16) bf16 Bs[2][128 * LDK];
  const int tid  = threadIdx.x;
  const int lane = tid & 31;
  const int wave = tid >> 5;
  const int wm   = (wave & 3) * 32;
  const int wn   = (wave >> 2) * 64;
  const int half = lane >> 4;
  const int r16  = lane & 15;
  // wave-0 issues all TDM descriptors; scalar condition -> true scalar branch
  const bool issuer = (__builtin_amdgcn_readfirstlane(tid) >> 5) == 0;

  const unsigned asBase   = (unsigned)(unsigned long long)(const void*)&As[0][0];
  const unsigned bsBase   = (unsigned)(unsigned long long)(const void*)&Bs[0][0];
  const unsigned bufBytes = 128u * LDK * sizeof(bf16);

  v8f acc[2][4];
  const v8f vzero = {0.f, 0.f, 0.f, 0.f, 0.f, 0.f, 0.f, 0.f};
#pragma unroll
  for (int i = 0; i < 2; i++)
#pragma unroll
    for (int j = 0; j < 4; j++) acc[i][j] = vzero;

  const int kSteps = Kdim >> 5;
  if (issuer) {
    tdm_load_tile_128x32(asBase, A + (size_t)bm * lda, lda);
    tdm_load_tile_128x32(bsBase, Bt + (size_t)bn * ldb, ldb);
  }
  for (int kt = 0; kt < kSteps; ++kt) {
    __syncthreads();  // all waves done reading buffer (kt+1)&1 from iter kt-1
    if (issuer) {
      if (kt + 1 < kSteps) {
        const int kk = (kt + 1) << 5;
        const unsigned bo = (unsigned)((kt + 1) & 1) * bufBytes;
        tdm_load_tile_128x32(asBase + bo, A + (size_t)bm * lda + kk, lda);
        tdm_load_tile_128x32(bsBase + bo, Bt + (size_t)bn * ldb + kk, ldb);
        __builtin_amdgcn_s_wait_tensorcnt(2);  // tile kt landed; kt+1 in flight
      } else {
        __builtin_amdgcn_s_wait_tensorcnt(0);
      }
    }
    __syncthreads();  // tile kt visible to every wave
    const bf16* Ab = As[kt & 1];
    const bf16* Bb = Bs[kt & 1];
    v16bf af[2], bfb[4];
    af[0] = frag_a(Ab, wm + r16, half);
    af[1] = frag_a(Ab, wm + 16 + r16, half);
#pragma unroll
    for (int j = 0; j < 4; j++) bfb[j] = frag_b(Bb, wn + j * 16 + r16, half);
#pragma unroll
    for (int i = 0; i < 2; i++)
#pragma unroll
      for (int j = 0; j < 4; j++)
        acc[i][j] = __builtin_amdgcn_wmma_f32_16x16x32_bf16(
            false, af[i], false, bfb[j], (short)0, acc[i][j], false, false);
  }
  epilogue<OUT_BF16>(acc, C, ldc, scale, bm, bn, wm, wn, half, r16);
}

// ---------------- GEMM core, B transposed in LDS (for P*V) -----------------
// Bm is B row-major [K x N]; A staged by TDM, B staged manually with transpose.
__device__ __forceinline__ void gemm_core_transB(const bf16* __restrict__ A, int lda,
                                                 const bf16* __restrict__ Bm, int ldb,
                                                 float* __restrict__ C, int ldc, int Kdim,
                                                 float scale, int bm, int bn) {
  __shared__ __align__(16) bf16 As[128 * LDK];
  __shared__ __align__(16) bf16 Bs[128 * LDK];  // Bs[n][k]
  const int tid  = threadIdx.x;
  const int lane = tid & 31;
  const int wave = tid >> 5;
  const int wm   = (wave & 3) * 32;
  const int wn   = (wave >> 2) * 64;
  const int half = lane >> 4;
  const int r16  = lane & 15;
  const bool issuer = (__builtin_amdgcn_readfirstlane(tid) >> 5) == 0;
  const unsigned asBase = (unsigned)(unsigned long long)(const void*)&As[0];

  v8f acc[2][4];
  const v8f vzero = {0.f, 0.f, 0.f, 0.f, 0.f, 0.f, 0.f, 0.f};
#pragma unroll
  for (int i = 0; i < 2; i++)
#pragma unroll
    for (int j = 0; j < 4; j++) acc[i][j] = vzero;

  const int kSteps = Kdim >> 5;
  for (int kt = 0; kt < kSteps; ++kt) {
    const int kk = kt << 5;
    __syncthreads();
    if (issuer) tdm_load_tile_128x32(asBase, A + (size_t)bm * lda + kk, lda);
    {
      // B tile [32 k x 128 n]: coalesced global read, transposed LDS store
      const int krow = tid & 31;
      const int nc   = (tid >> 5) * 16;
      const v8bf* gb = (const v8bf*)(Bm + (size_t)(kk + krow) * ldb + bn + nc);
      v8bf b0 = gb[0], b1 = gb[1];
#pragma unroll
      for (int j = 0; j < 8; j++) Bs[(nc + j) * LDK + krow]     = b0[j];
#pragma unroll
      for (int j = 0; j < 8; j++) Bs[(nc + 8 + j) * LDK + krow] = b1[j];
    }
    if (issuer) __builtin_amdgcn_s_wait_tensorcnt(0);
    __syncthreads();

    v16bf af[2], bfb[4];
    af[0] = frag_a(As, wm + r16, half);
    af[1] = frag_a(As, wm + 16 + r16, half);
#pragma unroll
    for (int j = 0; j < 4; j++) bfb[j] = frag_b(Bs, wn + j * 16 + r16, half);
#pragma unroll
    for (int i = 0; i < 2; i++)
#pragma unroll
      for (int j = 0; j < 4; j++)
        acc[i][j] = __builtin_amdgcn_wmma_f32_16x16x32_bf16(
            false, af[i], false, bfb[j], (short)0, acc[i][j], false, false);
  }
  epilogue<false>(acc, (void*)C, ldc, scale, bm, bn, wm, wn, half, r16);
}

// ---------------- conversion kernels ----------------
__global__ __launch_bounds__(256) void k_cvt_x(const float* __restrict__ x,
                                               bf16* __restrict__ xb, int n4) {
  int i = blockIdx.x * blockDim.x + threadIdx.x;
  if (i >= n4) return;
  v4f v = ((const v4f*)x)[i];
  v4bf o;
  o[0] = (bf16)v[0]; o[1] = (bf16)v[1]; o[2] = (bf16)v[2]; o[3] = (bf16)v[3];
  ((v4bf*)xb)[i] = o;
}

// Wt[z][n][k] = W_z[k][n]
__global__ __launch_bounds__(256) void k_cvt_wT(const float* __restrict__ Wq,
                                                const float* __restrict__ Wk,
                                                const float* __restrict__ Wv,
                                                bf16* __restrict__ Wt) {
  int z = blockIdx.z;
  const float* W = (z == 0) ? Wq : ((z == 1) ? Wk : Wv);
  bf16* out = Wt + (size_t)z * D_ * D_;
  int o = blockIdx.x * 256 + threadIdx.x;
  int n = o >> 10;
  int k = o & (D_ - 1);
  out[o] = (bf16)W[(size_t)k * D_ + n];
}

// ---------------- GEMM kernels ----------------
__global__ __launch_bounds__(256, 1) void k_qkv(const bf16* __restrict__ xb,
                                                const bf16* __restrict__ Wt,
                                                bf16* __restrict__ Q, bf16* __restrict__ K,
                                                bf16* __restrict__ V) {
  const int z = blockIdx.z;
  bf16* out = (z == 0) ? Q : ((z == 1) ? K : V);
  const float scale = (z == 0) ? 0.03125f : 1.0f;  // 1/sqrt(1024) folded into Q
  gemm_core_tdm<true>(xb, D_, Wt + (size_t)z * D_ * D_, D_, out, D_, D_, scale,
                      blockIdx.y * 128, blockIdx.x * 128);
}

__global__ __launch_bounds__(256, 1) void k_scores(const bf16* __restrict__ Q,
                                                   const bf16* __restrict__ K,
                                                   bf16* __restrict__ P) {
  const int b = blockIdx.z;
  gemm_core_tdm<true>(Q + (size_t)b * S_ * D_, D_, K + (size_t)b * S_ * D_, D_,
                      P + (size_t)b * S_ * S_, S_, D_, 1.0f,
                      blockIdx.y * 128, blockIdx.x * 128);
}

__global__ __launch_bounds__(256) void k_softmax(bf16* __restrict__ P) {
  __shared__ float red[256];
  const int row = blockIdx.x;
  bf16* p = P + (size_t)row * S_;
  const int t = threadIdx.x;
  float v[8];
  float m = -3.4e38f;
#pragma unroll
  for (int j = 0; j < 8; j++) {
    v[j] = (float)p[t + 256 * j];
    m = fmaxf(m, v[j]);
  }
  red[t] = m;
  __syncthreads();
  for (int s = 128; s > 0; s >>= 1) {
    if (t < s) red[t] = fmaxf(red[t], red[t + s]);
    __syncthreads();
  }
  m = red[0];
  __syncthreads();
  float sum = 0.f;
#pragma unroll
  for (int j = 0; j < 8; j++) {
    v[j] = __expf(v[j] - m);
    sum += v[j];
  }
  red[t] = sum;
  __syncthreads();
  for (int s = 128; s > 0; s >>= 1) {
    if (t < s) red[t] += red[t + s];
    __syncthreads();
  }
  const float inv = 1.0f / red[0];
#pragma unroll
  for (int j = 0; j < 8; j++) p[t + 256 * j] = (bf16)(v[j] * inv);
}

__global__ __launch_bounds__(256, 1) void k_out(const bf16* __restrict__ P,
                                                const bf16* __restrict__ V,
                                                float* __restrict__ O) {
  const int b = blockIdx.z;
  gemm_core_transB(P + (size_t)b * S_ * S_, S_, V + (size_t)b * S_ * D_, D_,
                   O + (size_t)b * S_ * D_, D_, S_, 1.0f,
                   blockIdx.y * 128, blockIdx.x * 128);
}

// ---------------- launcher ----------------
extern "C" void kernel_launch(void* const* d_in, const int* in_sizes, int n_in,
                              void* d_out, int out_size, void* d_ws, size_t ws_size,
                              hipStream_t stream) {
  const float* x  = (const float*)d_in[0];
  const float* Wq = (const float*)d_in[1];
  const float* Wk = (const float*)d_in[2];
  const float* Wv = (const float*)d_in[3];
  float* out = (float*)d_out;

  // workspace: xb 32M | Wt 6M | Q 32M | K 32M | V 32M | P 64M  (~198 MB)
  char* ws = (char*)d_ws;
  size_t off = 0;
  bf16* xb = (bf16*)(ws + off); off += (size_t)B_ * S_ * D_ * sizeof(bf16);
  bf16* Wt = (bf16*)(ws + off); off += 3ull * D_ * D_ * sizeof(bf16);
  bf16* Qb = (bf16*)(ws + off); off += (size_t)B_ * S_ * D_ * sizeof(bf16);
  bf16* Kb = (bf16*)(ws + off); off += (size_t)B_ * S_ * D_ * sizeof(bf16);
  bf16* Vb = (bf16*)(ws + off); off += (size_t)B_ * S_ * D_ * sizeof(bf16);
  bf16* Pb = (bf16*)(ws + off); off += (size_t)B_ * S_ * S_ * sizeof(bf16);
  (void)ws_size; (void)in_sizes; (void)n_in; (void)out_size;

  const int n4 = B_ * S_ * D_ / 4;
  k_cvt_x<<<dim3(n4 / 256), 256, 0, stream>>>(x, xb, n4);
  k_cvt_wT<<<dim3(D_ * D_ / 256, 1, 3), 256, 0, stream>>>(Wq, Wk, Wv, Wt);
  k_qkv<<<dim3(D_ / 128, (B_ * S_) / 128, 3), 256, 0, stream>>>(xb, Wt, Qb, Kb, Vb);
  k_scores<<<dim3(S_ / 128, S_ / 128, B_), 256, 0, stream>>>(Qb, Kb, Pb);
  k_softmax<<<dim3(B_ * S_), 256, 0, stream>>>(Pb);
  k_out<<<dim3(D_ / 128, S_ / 128, B_), 256, 0, stream>>>(Pb, Vb, out);
}
